// GNpoolMLP_60730837565915
// MI455X (gfx1250) — compile-verified, matrix operand order checked by reading, not run on hardware
//
#include <hip/hip_runtime.h>
#include <hip/hip_bf16.h>

// ---------------------------------------------------------------------------
// Types
// ---------------------------------------------------------------------------
typedef __bf16 bf16_t;
typedef __attribute__((ext_vector_type(8)))  __bf16 v8bf;
typedef __attribute__((ext_vector_type(16))) __bf16 v16bf;
typedef __attribute__((ext_vector_type(8)))  float  v8f;
typedef __attribute__((ext_vector_type(4)))  int    v4i;

#define N_NODES  50000
#define N_EDGES  800000
#define MSG_DIM  128
#define HIDDEN   300
#define HID_PAD  320
#define N_HID    128
#define N_GRAPHS 64

#define WBUF_ELEMS 10240   // one weight double-buffer half: 20 tiles * 512 bf16 (20KB)
#define NTHREADS   512     // 16 waves

// ---------------------------------------------------------------------------
// gfx1250 async global->LDS copy (ASYNCcnt) with sync fallback
// ---------------------------------------------------------------------------
#if __has_builtin(__builtin_amdgcn_global_load_async_to_lds_b128)
#define HAVE_ASYNC 1
typedef __attribute__((address_space(1))) v4i* gv4i_p;
typedef __attribute__((address_space(3))) v4i* sv4i_p;
#else
#define HAVE_ASYNC 0
#endif

__device__ __forceinline__ void async_wait_all() {
#if HAVE_ASYNC
#if __has_builtin(__builtin_amdgcn_s_wait_asynccnt)
    __builtin_amdgcn_s_wait_asynccnt(0);
#else
    asm volatile("s_wait_asynccnt 0x0" ::: "memory");
#endif
#endif
}

// Copy one K-chunk (NT tiles of 1KB, tile nt at global offset (nt*KT+kt)*1KB)
// into LDS Wbuf (tile nt at nt*1KB). 16B per lane per issue.
template<int KT, int NT>
__device__ __forceinline__ void stage_chunk(const bf16_t* __restrict__ Wp,
                                            bf16_t* __restrict__ Wbuf,
                                            int kt, int tid) {
    constexpr int CH = NT * 64;   // number of 16B chunks
    for (int j = tid; j < CH; j += NTHREADS) {
        int nt = j >> 6;
        int w  = j & 63;
        const char* s = (const char*)Wp + (((size_t)nt * KT + kt) << 10) + ((size_t)w << 4);
        char* d = (char*)Wbuf + (((size_t)nt) << 10) + ((size_t)w << 4);
#if HAVE_ASYNC
        __builtin_amdgcn_global_load_async_to_lds_b128(
            (gv4i_p)(s), (sv4i_p)(d), 0, 0);
#else
        *(uint4*)d = *(const uint4*)s;
#endif
    }
}

// ---------------------------------------------------------------------------
// WMMA helpers (bf16 operands, fp32 accumulate) — 16x16x32
// ---------------------------------------------------------------------------
__device__ __forceinline__ v8f wmma_bf16(v16bf a, v16bf b, v8f c) {
    return __builtin_amdgcn_wmma_f32_16x16x32_bf16(
        false, a, false, b, (short)0, c, false, false);
}

// A fragment (16x32, 16-bit) from row-major LDS tile.
__device__ __forceinline__ v16bf load_frag_a_lds(const bf16_t* base, int strideK,
                                                 int mrow0, int kt, int lane) {
    int m   = mrow0 + (lane & 15);
    int sel = lane >> 4;
    const bf16_t* p = base + m * strideK + kt * 32 + sel * 8;
    v8bf lo = *(const v8bf*)p;
    v8bf hi = *(const v8bf*)(p + 16);
    return __builtin_shufflevector(lo, hi, 0,1,2,3,4,5,6,7,8,9,10,11,12,13,14,15);
}

// B fragment (32x16, 16-bit) from an LDS-staged packed tile (column-contiguous).
__device__ __forceinline__ v16bf load_frag_b_lds(const bf16_t* tile, int lane) {
    const bf16_t* p = tile + (lane & 15) * 32 + (lane >> 4) * 8;
    v8bf lo = *(const v8bf*)p;
    v8bf hi = *(const v8bf*)(p + 16);
    return __builtin_shufflevector(lo, hi, 0,1,2,3,4,5,6,7,8,9,10,11,12,13,14,15);
}

// ---------------------------------------------------------------------------
// Fused MLP layer with double-buffered async weight staging.
// 16 waves: wave = 4 M-tiles x 4 N-stripes of NTW tiles each.
// ---------------------------------------------------------------------------
template<int KT, int NT, int NTW, bool RELU>
__device__ __forceinline__ void mlp_layer(const bf16_t* ldsIn, int strideIn,
                                          bf16_t* ldsOut, int strideOut,
                                          bf16_t* Wbuf,
                                          const bf16_t* __restrict__ Wp,
                                          const float* __restrict__ bias,
                                          int mrow0, int n0tile, int lane, int tid) {
    __syncthreads();                         // previous layer fully retired
    stage_chunk<KT, NT>(Wp, Wbuf, 0, tid);   // prologue: chunk 0 -> buf0

    v8f acc[NTW];
#pragma unroll
    for (int i = 0; i < NTW; ++i) acc[i] = (v8f){0.f,0.f,0.f,0.f,0.f,0.f,0.f,0.f};

    for (int kt = 0; kt < KT; ++kt) {
        async_wait_all();                    // my staging of chunk kt done
        __syncthreads();                     // everyone's staging done; buffers free
        if (kt + 1 < KT)
            stage_chunk<KT, NT>(Wp, Wbuf + ((kt + 1) & 1) * WBUF_ELEMS, kt + 1, tid);
        const bf16_t* cur = Wbuf + (kt & 1) * WBUF_ELEMS;
        v16bf a = load_frag_a_lds(ldsIn, strideIn, mrow0, kt, lane);
#pragma unroll
        for (int nt = 0; nt < NTW; ++nt) {
            v16bf b = load_frag_b_lds(cur + (size_t)(n0tile + nt) * 512, lane);
            acc[nt] = wmma_bf16(a, b, acc[nt]);
        }
    }

    int sel = lane >> 4;
    int nl  = lane & 15;
#pragma unroll
    for (int nt = 0; nt < NTW; ++nt) {
        int col = (n0tile + nt) * 16 + nl;
        float bv = bias[col];
#pragma unroll
        for (int r = 0; r < 8; ++r) {
            float v = acc[nt][r] + bv;
            if (RELU) v = fmaxf(v, 0.f);
            int row = mrow0 + r + sel * 8;
            ldsOut[row * strideOut + col] = (bf16_t)v;
        }
    }
}

// Final layer: global_atomic_add_f32 scatter into [*,128] via per-row index.
template<int KT, int NT, int NTW>
__device__ __forceinline__ void mlp_layer_out(const bf16_t* ldsIn, int strideIn,
                                              bf16_t* Wbuf,
                                              const bf16_t* __restrict__ Wp,
                                              const float* __restrict__ bias,
                                              float* __restrict__ outBase,
                                              const int* s_rowidx,
                                              int mrow0, int n0tile, int lane, int tid) {
    __syncthreads();
    stage_chunk<KT, NT>(Wp, Wbuf, 0, tid);

    v8f acc[NTW];
#pragma unroll
    for (int i = 0; i < NTW; ++i) acc[i] = (v8f){0.f,0.f,0.f,0.f,0.f,0.f,0.f,0.f};

    for (int kt = 0; kt < KT; ++kt) {
        async_wait_all();
        __syncthreads();
        if (kt + 1 < KT)
            stage_chunk<KT, NT>(Wp, Wbuf + ((kt + 1) & 1) * WBUF_ELEMS, kt + 1, tid);
        const bf16_t* cur = Wbuf + (kt & 1) * WBUF_ELEMS;
        v16bf a = load_frag_a_lds(ldsIn, strideIn, mrow0, kt, lane);
#pragma unroll
        for (int nt = 0; nt < NTW; ++nt) {
            v16bf b = load_frag_b_lds(cur + (size_t)(n0tile + nt) * 512, lane);
            acc[nt] = wmma_bf16(a, b, acc[nt]);
        }
    }

    int sel = lane >> 4;
    int nl  = lane & 15;
#pragma unroll
    for (int nt = 0; nt < NTW; ++nt) {
        int col = (n0tile + nt) * 16 + nl;
        float bv = bias[col];
#pragma unroll
        for (int r = 0; r < 8; ++r) {
            float v = acc[nt][r] + bv;
            int row = mrow0 + r + sel * 8;
            int g = s_rowidx[row];
            if (g >= 0) atomicAdd(outBase + (size_t)g * 128 + col, v);
        }
    }
}

// ---------------------------------------------------------------------------
// Edge message kernel: 64 edges/WG, fused 288->320->320->128
// ---------------------------------------------------------------------------
__global__ __launch_bounds__(NTHREADS)
void edge_msg_kernel(const bf16_t* __restrict__ xb, const float* __restrict__ ea,
                     const int* __restrict__ srcIdx, const int* __restrict__ dstIdx,
                     const bf16_t* __restrict__ W1p, const float* __restrict__ b1,
                     const bf16_t* __restrict__ W2p, const float* __restrict__ b2,
                     const bf16_t* __restrict__ W3p, const float* __restrict__ b3,
                     float* __restrict__ aggr) {
    __shared__ __align__(16) bf16_t bufA[64 * 320];
    __shared__ __align__(16) bf16_t bufB[64 * 320];
    __shared__ __align__(16) bf16_t bufW[2 * WBUF_ELEMS];
    __shared__ int s_dst[64];

    int tid  = threadIdx.x;
    int lane = tid & 31;
    int wave = tid >> 5;                 // 0..15
    int e0   = blockIdx.x * 64;

    // Gather [x_i | x_j | edge_attr] into LDS bf16, stride 288
    {
        int r = tid >> 3;                // 0..63
        int sub = tid & 7;               // 0..7, 16 cols each
        int e = e0 + r;
        int d = dstIdx[e];
        int s = srcIdx[e];
        if (sub == 0) s_dst[r] = d;

        const uint4* pd = (const uint4*)(xb + (size_t)d * 128 + sub * 16);
        const uint4* ps = (const uint4*)(xb + (size_t)s * 128 + sub * 16);
        uint4* qi = (uint4*)(bufA + r * 288 + sub * 16);
        uint4* qj = (uint4*)(bufA + r * 288 + 128 + sub * 16);
        qi[0] = pd[0]; qi[1] = pd[1];
        qj[0] = ps[0]; qj[1] = ps[1];

        if (sub < 4) {
            __align__(16) bf16_t tmp[8];
#pragma unroll
            for (int i = 0; i < 8; ++i)
                tmp[i] = (bf16_t)ea[(size_t)e * 32 + sub * 8 + i];
            *(uint4*)(bufA + r * 288 + 256 + sub * 8) = *(const uint4*)tmp;
        }
    }

    int mrow0 = (wave & 3) * 16;
    int q     = wave >> 2;               // N-stripe 0..3

    mlp_layer<9, 20, 5, true>(bufA, 288, bufB, 320, bufW, W1p, b1, mrow0, q * 5, lane, tid);
    mlp_layer<10, 20, 5, true>(bufB, 320, bufA, 320, bufW, W2p, b2, mrow0, q * 5, lane, tid);
    mlp_layer_out<10, 8, 2>(bufA, 320, bufW, W3p, b3, aggr, s_dst, mrow0, q * 2, lane, tid);
}

// ---------------------------------------------------------------------------
// Node update kernel: 64 nodes/WG, fused 256->320->320->128 -> graph sums
// ---------------------------------------------------------------------------
__global__ __launch_bounds__(NTHREADS)
void node_kernel(const bf16_t* __restrict__ xb, const float* __restrict__ aggr,
                 const int* __restrict__ batch,
                 const bf16_t* __restrict__ W1p, const float* __restrict__ b1,
                 const bf16_t* __restrict__ W2p, const float* __restrict__ b2,
                 const bf16_t* __restrict__ W3p, const float* __restrict__ b3,
                 float* __restrict__ gsum, int nNodes) {
    __shared__ __align__(16) bf16_t bufA[64 * 320];
    __shared__ __align__(16) bf16_t bufB[64 * 320];
    __shared__ __align__(16) bf16_t bufW[2 * WBUF_ELEMS];
    __shared__ int s_g[64];

    int tid  = threadIdx.x;
    int lane = tid & 31;
    int wave = tid >> 5;
    int n0   = blockIdx.x * 64;

    // Gather [x | aggr] into LDS bf16, stride 256
    {
        int r = tid >> 3;
        int sub = tid & 7;
        int node = n0 + r;
        int nc = (node < nNodes) ? node : (nNodes - 1);
        if (sub == 0) s_g[r] = (node < nNodes) ? batch[node] : -1;

        const uint4* px = (const uint4*)(xb + (size_t)nc * 128 + sub * 16);
        uint4* qx = (uint4*)(bufA + r * 256 + sub * 16);
        qx[0] = px[0]; qx[1] = px[1];

        const float* pa = aggr + (size_t)nc * 128 + sub * 16;
        __align__(16) bf16_t tmp[16];
#pragma unroll
        for (int i = 0; i < 16; ++i) tmp[i] = (bf16_t)pa[i];
        uint4* qa = (uint4*)(bufA + r * 256 + 128 + sub * 16);
        qa[0] = ((const uint4*)tmp)[0];
        qa[1] = ((const uint4*)tmp)[1];
    }

    int mrow0 = (wave & 3) * 16;
    int q     = wave >> 2;

    mlp_layer<8, 20, 5, true>(bufA, 256, bufB, 320, bufW, W1p, b1, mrow0, q * 5, lane, tid);
    mlp_layer<10, 20, 5, true>(bufB, 320, bufA, 320, bufW, W2p, b2, mrow0, q * 5, lane, tid);
    mlp_layer_out<10, 8, 2>(bufA, 320, bufW, W3p, b3, gsum, s_g, mrow0, q * 2, lane, tid);
}

// ---------------------------------------------------------------------------
// Small helper kernels
// ---------------------------------------------------------------------------
__global__ void zero_f32_kernel(float* p, size_t n) {
    size_t i = (size_t)blockIdx.x * 256 + threadIdx.x;
    if (i < n) p[i] = 0.f;
}

__global__ void cvt_x_kernel(const float* __restrict__ x, bf16_t* __restrict__ xb, size_t n) {
    size_t i = (size_t)blockIdx.x * 256 + threadIdx.x;
    if (i < n) xb[i] = (bf16_t)x[i];
}

// fp32 [fanin x fanout] row-major -> bf16 WMMA-B tiles (zero padded).
__global__ void pack_weight_kernel(const float* __restrict__ W, bf16_t* __restrict__ Wp,
                                   int fanin, int fanout, int finPad, int foutPad) {
    int KT = finPad >> 5;
    int NT = foutPad >> 4;
    size_t total = (size_t)KT * NT * 512;
    size_t i = (size_t)blockIdx.x * 256 + threadIdx.x;
    if (i >= total) return;
    int t  = (int)(i >> 9);
    int w  = (int)(i & 511);
    int nl = w >> 5;
    int kl = w & 31;
    int kt = t % KT;
    int nt = t / KT;
    int k = kt * 32 + kl;
    int n = nt * 16 + nl;
    float v = (k < fanin && n < fanout) ? W[(size_t)k * fanout + n] : 0.f;
    Wp[i] = (bf16_t)v;
}

__global__ void pad_bias_kernel(const float* __restrict__ src, float* __restrict__ dst,
                                int n, int npad) {
    int i = blockIdx.x * 256 + threadIdx.x;
    if (i < npad) dst[i] = (i < n) ? src[i] : 0.f;
}

__global__ void count_kernel(const int* __restrict__ batch, float* __restrict__ cnt, int n) {
    int i = blockIdx.x * 256 + threadIdx.x;
    if (i < n) atomicAdd(&cnt[batch[i]], 1.0f);
}

// Global MLP on pooled graph features (tiny, fp32 scalar)
__global__ __launch_bounds__(256)
void global_mlp_kernel(const float* __restrict__ gsum, const float* __restrict__ cnt,
                       const float* __restrict__ W1, const float* __restrict__ b1,
                       const float* __restrict__ W2, const float* __restrict__ b2,
                       const float* __restrict__ W3, const float* __restrict__ b3,
                       float* __restrict__ out) {
    __shared__ float hA[N_GRAPHS * 128];
    __shared__ float hB[N_GRAPHS * 128];
    int tid = threadIdx.x;

    for (int i = tid; i < N_GRAPHS * 128; i += 256) {
        int g = i >> 7;
        hA[i] = gsum[i] / fmaxf(cnt[g], 1.0f);
    }
    __syncthreads();
    for (int i = tid; i < N_GRAPHS * 128; i += 256) {
        int g = i >> 7, j = i & 127;
        float s = b1[j];
        for (int k = 0; k < 128; ++k) s += hA[g * 128 + k] * W1[k * 128 + j];
        hB[i] = fmaxf(s, 0.f);
    }
    __syncthreads();
    for (int i = tid; i < N_GRAPHS * 128; i += 256) {
        int g = i >> 7, j = i & 127;
        float s = b2[j];
        for (int k = 0; k < 128; ++k) s += hB[g * 128 + k] * W2[k * 128 + j];
        hA[i] = fmaxf(s, 0.f);
    }
    __syncthreads();
    if (tid < N_GRAPHS) {
        float s = b3[0];
        for (int k = 0; k < 128; ++k) s += hA[tid * 128 + k] * W3[k];
        out[tid] = s;
    }
}

// ---------------------------------------------------------------------------
// Launch
// ---------------------------------------------------------------------------
extern "C" void kernel_launch(void* const* d_in, const int* in_sizes, int n_in,
                              void* d_out, int out_size, void* d_ws, size_t ws_size,
                              hipStream_t stream) {
    (void)in_sizes; (void)n_in; (void)out_size; (void)ws_size;

    const float* x          = (const float*)d_in[0];
    const int*   edge_index = (const int*)d_in[1];
    const float* edge_attr  = (const float*)d_in[2];
    const int*   batch      = (const int*)d_in[3];
    const float* msg_W1 = (const float*)d_in[4];  const float* msg_b1 = (const float*)d_in[5];
    const float* msg_W2 = (const float*)d_in[6];  const float* msg_b2 = (const float*)d_in[7];
    const float* msg_W3 = (const float*)d_in[8];  const float* msg_b3 = (const float*)d_in[9];
    const float* node_W1 = (const float*)d_in[10]; const float* node_b1 = (const float*)d_in[11];
    const float* node_W2 = (const float*)d_in[12]; const float* node_b2 = (const float*)d_in[13];
    const float* node_W3 = (const float*)d_in[14]; const float* node_b3 = (const float*)d_in[15];
    const float* glob_W1 = (const float*)d_in[16]; const float* glob_b1 = (const float*)d_in[17];
    const float* glob_W2 = (const float*)d_in[18]; const float* glob_b2 = (const float*)d_in[19];
    const float* glob_W3 = (const float*)d_in[20]; const float* glob_b3 = (const float*)d_in[21];

    const int* srcIdx = edge_index;
    const int* dstIdx = edge_index + N_EDGES;

    char* ws = (char*)d_ws;
    size_t off = 0;
    auto carve = [&](size_t bytes) -> char* {
        char* p = ws + off;
        off = (off + bytes + 255) & ~(size_t)255;
        return p;
    };
    bf16_t* xb   = (bf16_t*)carve((size_t)N_NODES * 128 * sizeof(bf16_t));
    float*  aggr = (float*) carve((size_t)N_NODES * 128 * sizeof(float));
    float*  gsum = (float*) carve((size_t)N_GRAPHS * 128 * sizeof(float));
    float*  cnt  = (float*) carve((size_t)N_GRAPHS * sizeof(float));
    const size_t mW1n = 9 * 20 * 512, mW2n = 10 * 20 * 512, mW3n = 10 * 8 * 512;
    const size_t nW1n = 8 * 20 * 512, nW2n = 10 * 20 * 512, nW3n = 10 * 8 * 512;
    bf16_t* mW1p = (bf16_t*)carve(mW1n * sizeof(bf16_t));
    bf16_t* mW2p = (bf16_t*)carve(mW2n * sizeof(bf16_t));
    bf16_t* mW3p = (bf16_t*)carve(mW3n * sizeof(bf16_t));
    bf16_t* nW1p = (bf16_t*)carve(nW1n * sizeof(bf16_t));
    bf16_t* nW2p = (bf16_t*)carve(nW2n * sizeof(bf16_t));
    bf16_t* nW3p = (bf16_t*)carve(nW3n * sizeof(bf16_t));
    float* mb1p = (float*)carve(HID_PAD * sizeof(float));
    float* mb2p = (float*)carve(HID_PAD * sizeof(float));
    float* nb1p = (float*)carve(HID_PAD * sizeof(float));
    float* nb2p = (float*)carve(HID_PAD * sizeof(float));

    auto gblk = [](size_t n) { return (unsigned)((n + 255) / 256); };

    zero_f32_kernel<<<gblk((size_t)N_NODES * 128), 256, 0, stream>>>(aggr, (size_t)N_NODES * 128);
    zero_f32_kernel<<<gblk(N_GRAPHS * 128), 256, 0, stream>>>(gsum, N_GRAPHS * 128);
    zero_f32_kernel<<<1, 256, 0, stream>>>(cnt, N_GRAPHS);

    cvt_x_kernel<<<gblk((size_t)N_NODES * 128), 256, 0, stream>>>(x, xb, (size_t)N_NODES * 128);

    pack_weight_kernel<<<gblk(mW1n), 256, 0, stream>>>(msg_W1, mW1p, 288, HIDDEN, 288, HID_PAD);
    pack_weight_kernel<<<gblk(mW2n), 256, 0, stream>>>(msg_W2, mW2p, HIDDEN, HIDDEN, HID_PAD, HID_PAD);
    pack_weight_kernel<<<gblk(mW3n), 256, 0, stream>>>(msg_W3, mW3p, HIDDEN, MSG_DIM, HID_PAD, 128);
    pack_weight_kernel<<<gblk(nW1n), 256, 0, stream>>>(node_W1, nW1p, 256, HIDDEN, 256, HID_PAD);
    pack_weight_kernel<<<gblk(nW2n), 256, 0, stream>>>(node_W2, nW2p, HIDDEN, HIDDEN, HID_PAD, HID_PAD);
    pack_weight_kernel<<<gblk(nW3n), 256, 0, stream>>>(node_W3, nW3p, HIDDEN, N_HID, HID_PAD, 128);

    pad_bias_kernel<<<2, 256, 0, stream>>>(msg_b1, mb1p, HIDDEN, HID_PAD);
    pad_bias_kernel<<<2, 256, 0, stream>>>(msg_b2, mb2p, HIDDEN, HID_PAD);
    pad_bias_kernel<<<2, 256, 0, stream>>>(node_b1, nb1p, HIDDEN, HID_PAD);
    pad_bias_kernel<<<2, 256, 0, stream>>>(node_b2, nb2p, HIDDEN, HID_PAD);

    count_kernel<<<gblk(N_NODES), 256, 0, stream>>>(batch, cnt, N_NODES);

    edge_msg_kernel<<<N_EDGES / 64, NTHREADS, 0, stream>>>(
        xb, edge_attr, srcIdx, dstIdx,
        mW1p, mb1p, mW2p, mb2p, mW3p, msg_b3, aggr);

    node_kernel<<<(N_NODES + 63) / 64, NTHREADS, 0, stream>>>(
        xb, aggr, batch,
        nW1p, nb1p, nW2p, nb2p, nW3p, node_b3, gsum, N_NODES);

    global_mlp_kernel<<<1, 256, 0, stream>>>(
        gsum, cnt, glob_W1, glob_b1, glob_W2, glob_b2, glob_W3, glob_b3,
        (float*)d_out);
}
